// LocalAttention_47304769798205
// MI455X (gfx1250) — matrix operand order repaired and checked by reading, not compiled, not run on hardware
//
#include <hip/hip_runtime.h>

// CDNA5 wave32 WMMA types
typedef __attribute__((ext_vector_type(16))) _Float16 v16h;
typedef __attribute__((ext_vector_type(8)))  float    v8f;

#define NEG_BIG (-1.0e9f)

// ds_swizzle group-of-32 mode: offset = (xor<<10)|(or<<5)|and ; SWAPX16 = 0x401F
#define SWZ_X16 0x401F
template<int OFF>
static __device__ __forceinline__ float swz(float x) {
    int r = __builtin_amdgcn_ds_swizzle(__builtin_bit_cast(int, x), OFF);
    return __builtin_bit_cast(float, r);
}

// Problem constants from the reference setup
constexpr int Bn = 2, Tn = 2048, Hn = 8, En = 64;   // C = 128 (window = [t-64, t+63])

__global__ __launch_bounds__(32)
void local_attn_wmma_kernel(const float* __restrict__ Q,
                            const float* __restrict__ K,
                            const float* __restrict__ V,
                            float* __restrict__ O)
{
    const int lane = threadIdx.x;    // wave32
    const int lo   = lane & 15;
    const int hi   = lane >> 4;      // 0 or 1

    const int tilesPerBH = Tn / 16;
    const int tile = blockIdx.x % tilesPerBH;
    const int bh   = blockIdx.x / tilesPerBH;
    const int head = bh % Hn;
    const int b    = bh / Hn;
    const int t0   = tile * 16;

    const float temp = 0.125f;               // 1/sqrt(E), E = 64
    const long  rowStride = (long)Hn * En;   // 512 floats between consecutive t
    const float* Qb = Q + ((long)b * Tn * Hn + head) * En;
    const float* Kb = K + ((long)b * Tn * Hn + head) * En;
    const float* Vb = V + ((long)b * Tn * Hn + head) * En;
    float*       Ob = O + ((long)b * Tn * Hn + head) * En;

    const int t = t0 + lo;  // query row owned by this lane (as a D/B column)

    // ---- Q as B-matrix (Q^T) fragments, temp folded in.
    // B layout: element j <-> (k = 32w + 16*hi + j, n = lane&15)  -> contiguous e per lane
    v16h qb[2];
    {
        const float* qrow = Qb + (long)t * rowStride;
        #pragma unroll
        for (int w = 0; w < 2; ++w)
            #pragma unroll
            for (int j = 0; j < 16; ++j)
                qb[w][j] = (_Float16)(qrow[32 * w + 16 * hi + j] * temp);
    }

    // O^T accumulators: acc[c] element r <-> (e = 16c + 8*hi + r, query t)
    v8f acc[4];
    #pragma unroll
    for (int c = 0; c < 4; ++c) acc[c] = (v8f){};
    float mrow = -3.0e38f;   // running row max
    float lrow = 0.0f;       // running row sum

    // 10 key tiles covering [t0-64, t0+96), processed as 5 pairs of 16
    for (int p = 0; p < 5; ++p) {
        const int pbase = t0 - 64 + 32 * p;

        // ---- S^T = K_tile * Q^T for the two sub-tiles of the pair
        v8f d01[2];
        #pragma unroll
        for (int st = 0; st < 2; ++st) {
            const int sb   = pbase + 16 * st;
            const int srow = sb + lo;
            const int sc   = min(max(srow, 0), Tn - 1);   // clamp; masked below
            const float* krow = Kb + (long)sc * rowStride;
            v8f d = (v8f){};
            #pragma unroll
            for (int w = 0; w < 2; ++w) {
                // A layout: element j <-> (m = lane&15, k = 16*hi + j)
                v16h ka;
                #pragma unroll
                for (int j = 0; j < 16; ++j) {
                    const int e = 32 * w + hi * 16 + j;
                    ka[j] = (_Float16)krow[e];
                }
                d = __builtin_amdgcn_wmma_f32_16x16x32_f16(
                        false, ka, false, qb[w], (short)0, d, false, false);
            }
            // mask: D element r -> key s = sb + 8*hi + r, query t
            #pragma unroll
            for (int r = 0; r < 8; ++r) {
                const int s  = sb + 8 * hi + r;
                const int dd = s - t;
                const bool valid = (s >= 0) && (s < Tn) && (dd >= -64) && (dd <= 63);
                d[r] = valid ? d[r] : NEG_BIG;
            }
            d01[st] = d;
        }

        // ---- online softmax: in-lane 16-elem reduce + one xor-16 swizzle
        float pm = NEG_BIG;
        #pragma unroll
        for (int r = 0; r < 8; ++r) pm = fmaxf(pm, fmaxf(d01[0][r], d01[1][r]));
        pm = fmaxf(pm, swz<SWZ_X16>(pm));
        const float mnew  = fmaxf(mrow, pm);
        const float alpha = __expf(mrow - mnew);

        v8f P0, P1;
        float psum = 0.0f;
        #pragma unroll
        for (int r = 0; r < 8; ++r) {
            P0[r] = __expf(d01[0][r] - mnew);
            P1[r] = __expf(d01[1][r] - mnew);
            psum += P0[r] + P1[r];
        }
        psum += swz<SWZ_X16>(psum);
        lrow = lrow * alpha + psum;
        mrow = mnew;
        #pragma unroll
        for (int c = 0; c < 4; ++c)
            #pragma unroll
            for (int r = 0; r < 8; ++r) acc[c][r] *= alpha;

        // ---- P^T (D layout) -> B-matrix fragment (keys 0..31 x queries 0..15)
        // lane<16 sources sub-tile 0; lane>=16 sources sub-tile 1; partner via xor-16
        v16h pb;
        #pragma unroll
        for (int j = 0; j < 8; ++j) {
            const float a0 = swz<SWZ_X16>(P0[j]);   // partner lane's P0[j]
            const float a1 = swz<SWZ_X16>(P1[j]);   // partner lane's P1[j]
            pb[j]     = (_Float16)(hi ? a1 : P0[j]);   // element j   (k = 16*hi + j, key m = j)
            pb[j + 8] = (_Float16)(hi ? P1[j] : a0);   // element j+8 (key m = j+8)
        }

        // ---- O^T chunks: acc[c] += V^T_chunk(16e x 32keys) * P^T(32keys x 16q)
        #pragma unroll
        for (int c = 0; c < 4; ++c) {
            const int e = 16 * c + lo;   // A row m = lane&15 -> feature e
            v16h va;
            #pragma unroll
            for (int j = 0; j < 16; ++j) {
                const int k = hi * 16 + j;
                const int s = min(max(pbase + k, 0), Tn - 1);   // masked keys have P=0
                va[j] = (_Float16)Vb[(long)s * rowStride + e];
            }
            acc[c] = __builtin_amdgcn_wmma_f32_16x16x32_f16(
                        false, va, false, pb, (short)0, acc[c], false, false);
        }
    }

    // ---- epilogue: out[t][e] = acc / l ; e = 16c + 8*hi + r (contiguous runs of 8)
    const float inv = 1.0f / lrow;
    float* orow = Ob + (long)t * rowStride;
    #pragma unroll
    for (int c = 0; c < 4; ++c)
        #pragma unroll
        for (int r = 0; r < 8; ++r)
            orow[16 * c + 8 * hi + r] = acc[c][r] * inv;
}

extern "C" void kernel_launch(void* const* d_in, const int* in_sizes, int n_in,
                              void* d_out, int out_size, void* d_ws, size_t ws_size,
                              hipStream_t stream) {
    const float* q = (const float*)d_in[0];
    const float* k = (const float*)d_in[1];
    const float* v = (const float*)d_in[2];
    // d_in[3] = local_context (=128), baked into the kernel's window logic
    float* out = (float*)d_out;

    const int grid = Bn * Hn * (Tn / 16);   // 2048 waves, one per (b,h,q-tile)
    local_attn_wmma_kernel<<<grid, 32, 0, stream>>>(q, k, v, out);
}
